// SparseAttention_42451456753942
// MI455X (gfx1250) — matrix operand order repaired and checked by reading, not compiled, not run on hardware
//
#include <hip/hip_runtime.h>
#include <hip/hip_bf16.h>

typedef __attribute__((ext_vector_type(16))) _Float16 v16h;
typedef __attribute__((ext_vector_type(8)))  float    v8f;

#define N_CTX   2048
#define DMODEL  1024
#define HEADS   16
#define DHEAD   64
#define BAND    128
#define QBLK    32                  // queries per wave (2 x 16-query tiles)
#define QGRPS   (N_CTX / QBLK)      // 64

// 1/sqrt(64) * log2(e): scores computed directly in log2 domain so the
// hardware base-2 v_exp_f32 needs no per-element ln2 fixup.
#define QSCALE  0.18033688011112042f

// Base-2 exponential -> v_exp_f32
__device__ __forceinline__ float fast_exp2(float x) {
#if __has_builtin(__builtin_amdgcn_exp2f)
  return __builtin_amdgcn_exp2f(x);
#else
  return exp2f(x);
#endif
}

// Cross-half exchange: ds_swizzle SWAPX16 (group-of-32: xor=0x10, and=0x1f)
__device__ __forceinline__ float xor16(float x) {
  return __int_as_float(__builtin_amdgcn_ds_swizzle(__float_as_int(x), 0x401F));
}

__global__ __launch_bounds__(32)
void sparse_attn_wmma_kernel(const float* __restrict__ Q,
                             const float* __restrict__ K,
                             const float* __restrict__ V,
                             float* __restrict__ Out) {
  const int lane = threadIdx.x;       // 0..31 (wave32)
  const int half = lane >> 4;         // 0/1 lane group
  const int l16  = lane & 15;
  const int hs   = half * 8;          // accumulator row offset per lane group

  const int bid  = blockIdx.x;
  const int h    = bid >> 6;          // / QGRPS
  const int qg   = bid & (QGRPS - 1);
  const int q0   = qg * QBLK;
  const int hoff = h * DHEAD;

  // ---- B operands for S^T = K * Q^T (Q pre-scaled into log2 domain) -------
  // B layout (32x16, f16): element e -> K-dim index = 16*half + e, N = l16.
  v16h b_q[2][2];
  #pragma unroll
  for (int qt = 0; qt < 2; ++qt) {
    const float* qrow = Q + (size_t)(q0 + qt * 16 + l16) * DMODEL + hoff;
    #pragma unroll
    for (int c = 0; c < 2; ++c) {
      const float* p = qrow + c * 32 + half * 16;
      #pragma unroll
      for (int e = 0; e < 16; ++e)
        b_q[qt][c][e] = (_Float16)(p[e] * QSCALE);
    }
  }

  // O^T accumulators: per query tile, 4 chunks; chunk c holds
  // d = c*16 + r + 8*half, query = l16
  v8f acc[2][4] = {};
  float m[2]    = {-1e30f, -1e30f};   // running max (log2 domain, replicated)
  float lsum[2] = {0.0f, 0.0f};       // running denom

  int jstart = q0 - BAND;        if (jstart < 0)   jstart = 0;
  int jend   = q0 + QBLK + BAND; if (jend > N_CTX) jend   = N_CTX;

  for (int j0 = jstart; j0 < jend; j0 += 32) {
    // wave-uniform: does this 32-key step need masking / row clamping?
    const bool edge = (j0 < q0 - 96) | (j0 > q0 + 96) | (j0 + 32 > N_CTX);

    // ---- A operands: two 16-key tiles of K, each split into two K=32 chunks
    // A layout (16x32, f16): M = l16 (key row), element e -> K-dim d:
    //   d = chunk*32 + 16*half + e
    v16h a_k[2][2];
    #pragma unroll
    for (int t = 0; t < 2; ++t) {
      int row = j0 + t * 16 + l16;
      if (edge) { if (row > N_CTX - 1) row = N_CTX - 1; }   // clamp (masked anyway)
      const float* kr = K + (size_t)row * DMODEL + hoff;
      #pragma unroll
      for (int c = 0; c < 2; ++c) {
        #pragma unroll
        for (int e = 0; e < 16; ++e) {
          int d = c * 32 + 2 * hs + e;
          a_k[t][c][e] = (_Float16)kr[d];
        }
      }
    }

    // ---- S^T tiles: s[qt][t] element r -> (key = j0+16t+r+hs, query = l16)
    v8f s[2][2];
    #pragma unroll
    for (int qt = 0; qt < 2; ++qt) {
      #pragma unroll
      for (int t = 0; t < 2; ++t) {
        v8f cz = {};
        cz = __builtin_amdgcn_wmma_f32_16x16x32_f16(false, a_k[t][0], false, b_q[qt][0],
                                                    (short)0, cz, false, false);
        s[qt][t] = __builtin_amdgcn_wmma_f32_16x16x32_f16(false, a_k[t][1], false, b_q[qt][1],
                                                          (short)0, cz, false, false);
      }
    }

    // ---- per query tile: (edge) mask, online softmax, build B = P^T -------
    v16h b_p[2];
    #pragma unroll
    for (int qt = 0; qt < 2; ++qt) {
      if (edge) {
        const int i_glob = q0 + qt * 16 + l16;
        #pragma unroll
        for (int t = 0; t < 2; ++t) {
          #pragma unroll
          for (int r = 0; r < 8; ++r) {
            int j = j0 + t * 16 + r + hs;
            int dist = i_glob - j; if (dist < 0) dist = -dist;
            bool ok = (dist <= BAND) && (j < N_CTX);
            if (!ok) s[qt][t][r] = -1e30f;
          }
        }
      }

      // online softmax in log2 domain (per query = l16; replicated halves)
      float stepmax = -1e30f;
      #pragma unroll
      for (int t = 0; t < 2; ++t)
        #pragma unroll
        for (int r = 0; r < 8; ++r)
          stepmax = fmaxf(stepmax, s[qt][t][r]);
      stepmax = fmaxf(stepmax, xor16(stepmax));

      float m_new = fmaxf(m[qt], stepmax);
      float corr  = fast_exp2(m[qt] - m_new);
      float psum  = 0.0f;
      #pragma unroll
      for (int t = 0; t < 2; ++t)
        #pragma unroll
        for (int r = 0; r < 8; ++r) {
          float p = fast_exp2(s[qt][t][r] - m_new);
          s[qt][t][r] = p;
          psum += p;
        }
      psum += xor16(psum);
      lsum[qt] = lsum[qt] * corr + psum;
      m[qt] = m_new;
      #pragma unroll
      for (int c = 0; c < 4; ++c)
        #pragma unroll
        for (int r = 0; r < 8; ++r)
          acc[qt][c][r] *= corr;

      // build B = P^T (32 keys x 16 queries) via SWAPX16
      // element e -> key = 16*half + e, N = l16
      #pragma unroll
      for (int r = 0; r < 8; ++r) {
        float s0o = xor16(s[qt][0][r]);   // partner's tile0 row
        float s1o = xor16(s[qt][1][r]);   // partner's tile1 row
        b_p[qt][r]     = (_Float16)(half ? s1o         : s[qt][0][r]);
        b_p[qt][8 + r] = (_Float16)(half ? s[qt][1][r] : s0o);
      }
    }

    // ---- O^T += V^T * P^T : A = V^T chunk shared by both query tiles ------
    #pragma unroll
    for (int c = 0; c < 4; ++c) {
      v16h a_v;
      #pragma unroll
      for (int e = 0; e < 16; ++e) {
        int k = 2 * hs + e;
        int row = j0 + k;
        if (edge) { if (row > N_CTX - 1) row = N_CTX - 1; }
        a_v[e] = (_Float16)V[(size_t)row * DMODEL + hoff + c * 16 + l16];
      }
      #pragma unroll
      for (int qt = 0; qt < 2; ++qt)
        acc[qt][c] = __builtin_amdgcn_wmma_f32_16x16x32_f16(false, a_v, false, b_p[qt],
                                                            (short)0, acc[qt][c],
                                                            false, false);
    }
  }

  // ---- normalize + store: Out[(q0+qt*16+l16)*D + hoff + c*16 + r + 8*half]
  #pragma unroll
  for (int qt = 0; qt < 2; ++qt) {
    float inv = 1.0f / lsum[qt];
    float* orow = Out + (size_t)(q0 + qt * 16 + l16) * DMODEL + hoff;
    #pragma unroll
    for (int c = 0; c < 4; ++c)
      #pragma unroll
      for (int r = 0; r < 8; ++r)
        orow[c * 16 + r + 8 * half] = acc[qt][c][r] * inv;
  }
}

extern "C" void kernel_launch(void* const* d_in, const int* in_sizes, int n_in,
                              void* d_out, int out_size, void* d_ws, size_t ws_size,
                              hipStream_t stream) {
  const float* q = (const float*)d_in[0];
  const float* k = (const float*)d_in[1];
  const float* v = (const float*)d_in[2];
  float* out = (float*)d_out;
  (void)in_sizes; (void)n_in; (void)out_size; (void)d_ws; (void)ws_size;

  dim3 grid(HEADS * QGRPS);  // 1024 blocks
  dim3 block(32);            // one wave32 per block
  sparse_attn_wmma_kernel<<<grid, block, 0, stream>>>(q, k, v, out);
}
